// HadamardW4A4LinearFast_1176821039709
// MI455X (gfx1250) — 8-bit matrix operand order repaired and checked by reading, compiled, not run on hardware
//
#include <hip/hip_runtime.h>
#include <stdint.h>

typedef int v8i __attribute__((ext_vector_type(8)));
typedef int v4i __attribute__((ext_vector_type(4)));

#define N_TOKENS 8192
#define IN_F     4096
#define OUT_F    4096
#define KBLKS    (IN_F / 64)

#define BM 16
#define BN 16
#define TILE_BYTES 1024
#define A_BYTES (BM * TILE_BYTES)
#define B_BYTES (BN * TILE_BYTES)
#define BUF_BYTES (A_BYTES + B_BYTES)

#if defined(__has_builtin)
# if __has_builtin(__builtin_amdgcn_global_load_async_to_lds_b128)
#  define HAS_ASYNC_COPY 1
# endif
#endif
#ifndef HAS_ASYNC_COPY
# define HAS_ASYNC_COPY 0
#endif

#if HAS_ASYNC_COPY
typedef __attribute__((address_space(1))) v4i v4i_g;
typedef __attribute__((address_space(3))) v4i v4i_l;

__device__ __forceinline__ void async_cp16(const uint8_t* g, uint8_t* l) {
    __builtin_amdgcn_global_load_async_to_lds_b128(
        (v4i_g*)(uintptr_t)g, (v4i_l*)(uint32_t)(uintptr_t)l, 0, 0);
}
__device__ __forceinline__ void wait_async_le4() {
# if __has_builtin(__builtin_amdgcn_s_wait_asynccnt)
    __builtin_amdgcn_s_wait_asynccnt(4);
# else
    asm volatile("s_wait_asynccnt 4" ::: "memory");
# endif
}
__device__ __forceinline__ void wait_async_le0() {
# if __has_builtin(__builtin_amdgcn_s_wait_asynccnt)
    __builtin_amdgcn_s_wait_asynccnt(0);
# else
    asm volatile("s_wait_asynccnt 0" ::: "memory");
# endif
}
#endif

__device__ __forceinline__ int a_frag_pos(int r, int klocal) {
    int g = klocal >> 4;
    int w16 = klocal & 15;
    int h = w16 >> 3;
    int p = (w16 >> 2) & 1;
    int b = klocal & 3;
    int lane = r + h * 16;
    int v = g * 2 + p;
    return lane * 32 + v * 4 + b;
}

__device__ __forceinline__ int b_frag_pos(int c, int klocal) {
    int half32 = klocal >> 4;
    int w32 = klocal & 31;
    int h16 = (w32 >> 3) & 1;
    int q = (w32 >> 2) & 1;
    int b = klocal & 3;
    int lane = c + h16 * 16;
    int v = half32 * 2 + q;
    return lane * 32 + v * 4 + b;
}

__global__ __launch_bounds__(256) void quant_x_kernel(
    const float* __restrict__ x, uint8_t* __restrict__ xq,
    float* __restrict__ xs) {
    const int row = blockIdx.x;
    const int t = threadIdx.x;

    const float4* xr4 = (const float4*)(x + (size_t)row * IN_F);
    float vals[16];
    float m = 0.0f;
#pragma unroll
    for (int i = 0; i < 4; ++i) {
        float4 f = xr4[t * 4 + i];
        vals[i * 4 + 0] = f.x; vals[i * 4 + 1] = f.y;
        vals[i * 4 + 2] = f.z; vals[i * 4 + 3] = f.w;
        m = fmaxf(m, fmaxf(fmaxf(fabsf(f.x), fabsf(f.y)),
                           fmaxf(fabsf(f.z), fabsf(f.w))));
    }

    __shared__ float red[256];
    red[t] = m;
    __syncthreads();
#pragma unroll
    for (int s = 128; s > 0; s >>= 1) {
        if (t < s) red[t] = fmaxf(red[t], red[t + s]);
        __syncthreads();
    }
    const float scale = fmaxf(red[0] * (1.0f / 7.0f), 1e-5f);
    if (t == 0) xs[row] = scale;
    const float inv = 1.0f / scale;

    const int mtile = row >> 4;
    const int r = row & 15;
    uint8_t* base = xq + (size_t)mtile * KBLKS * 1024;

#pragma unroll
    for (int d = 0; d < 4; ++d) {
        const int k0 = t * 16 + d * 4;
        uint32_t pack = 0;
#pragma unroll
        for (int b = 0; b < 4; ++b) {
            float q = __builtin_rintf(vals[d * 4 + b] * inv);
            q = fminf(fmaxf(q, -8.0f), 7.0f);
            int qi = (int)q;
            pack |= ((uint32_t)(qi & 0xFF)) << (8 * b);
        }
        const int kb = k0 >> 6;
        const int kl = k0 & 63;
        *(uint32_t*)(base + (size_t)kb * 1024 + a_frag_pos(r, kl)) = pack;
    }
}

__global__ __launch_bounds__(256) void unpack_w_kernel(
    const int* __restrict__ wp, uint8_t* __restrict__ wq) {
    const int o = blockIdx.x;
    const int t = threadIdx.x;
    const int ntile = o >> 4;
    const int c = o & 15;
    const int* wr = wp + (size_t)o * (IN_F / 2);

#pragma unroll
    for (int d = 0; d < 4; ++d) {
        const int k4 = t * 4 + d;
        const int p0 = wr[k4 * 2 + 0];
        const int p1 = wr[k4 * 2 + 1];
        const int b0 = ((p0 & 15) ^ 8) - 8;
        const int b1 = (((p0 >> 4) & 15) ^ 8) - 8;
        const int b2 = ((p1 & 15) ^ 8) - 8;
        const int b3 = (((p1 >> 4) & 15) ^ 8) - 8;
        const uint32_t pack = (uint32_t)(b0 & 0xFF)
                            | ((uint32_t)(b1 & 0xFF) << 8)
                            | ((uint32_t)(b2 & 0xFF) << 16)
                            | ((uint32_t)(b3 & 0xFF) << 24);
        const int k = k4 * 4;
        const int kb = k >> 6;
        const int kl = k & 63;
        *(uint32_t*)(wq + ((size_t)ntile * KBLKS + kb) * 1024
                     + b_frag_pos(c, kl)) = pack;
    }
}

__global__ __launch_bounds__(512) void gemm_w4a4_kernel(
    const uint8_t* __restrict__ aq, const uint8_t* __restrict__ bq,
    const float* __restrict__ xs, const float* __restrict__ wscale,
    float* __restrict__ out) {
    __shared__ uint8_t smem[2 * BUF_BYTES];

    const int t = threadIdx.x;
    const int lane = t & 31;
    const int waveId = t >> 5;
    const int wm = waveId >> 2;
    const int wn = waveId & 3;

    const int bm = blockIdx.x / (OUT_F / 256);
    const int bn = blockIdx.x % (OUT_F / 256);

    const int chunk = waveId;
    const int soff = lane * 32;
    const uint8_t* gA = aq + (size_t)(bm * BM + chunk) * KBLKS * 1024 + soff;
    const uint8_t* gB = bq + (size_t)(bn * BN + chunk) * KBLKS * 1024 + soff;
    uint8_t* lA = smem + chunk * 1024 + soff;
    uint8_t* lB = smem + A_BYTES + chunk * 1024 + soff;

    auto stage = [&](int buf, int kb) {
        const uint8_t* ga = gA + (size_t)kb * 1024;
        const uint8_t* gb = gB + (size_t)kb * 1024;
        uint8_t* la = lA + buf * BUF_BYTES;
        uint8_t* lb = lB + buf * BUF_BYTES;
#if HAS_ASYNC_COPY
        async_cp16(ga, la);
        async_cp16(ga + 16, la + 16);
        async_cp16(gb, lb);
        async_cp16(gb + 16, lb + 16);
#else
        *(v8i*)la = *(const v8i*)ga;
        *(v8i*)lb = *(const v8i*)gb;
#endif
    };

    v8i acc[4][4] = {};
    const int aoff = wm * 4 * 1024 + lane * 32;
    const int boff = A_BYTES + wn * 4 * 1024 + lane * 32;

    auto compute = [&](int buf) {
        const uint8_t* base = smem + buf * BUF_BYTES;
        v8i a[4], b[4];
#pragma unroll
        for (int i = 0; i < 4; ++i)
            a[i] = *(const v8i*)(base + aoff + i * 1024);
#pragma unroll
        for (int j = 0; j < 4; ++j)
            b[j] = *(const v8i*)(base + boff + j * 1024);
#pragma unroll
        for (int i = 0; i < 4; ++i)
#pragma unroll
            for (int j = 0; j < 4; ++j)
                acc[i][j] = __builtin_amdgcn_wmma_i32_16x16x64_iu8(
                    true, a[i], true, b[j], acc[i][j], false, false);
    };

#if HAS_ASYNC_COPY
    stage(0, 0);
    stage(1, 1);
    for (int kb = 0; kb < KBLKS; ++kb) {
        if (kb < KBLKS - 1) wait_async_le4();
        else                wait_async_le0();
        __syncthreads();
        compute(kb & 1);
        __syncthreads();
        if (kb + 2 < KBLKS) stage(kb & 1, kb + 2);
    }
#else
    for (int kb = 0; kb < KBLKS; ++kb) {
        stage(0, kb);
        __syncthreads();
        compute(0);
        __syncthreads();
    }
#endif

    const int mt0 = bm * BM + wm * 4;
    const int nt0 = bn * BN + wn * 4;
    const int col0 = lane & 15;
    const int rowh = (lane >> 4) * 8;
    float wsc[4];
#pragma unroll
    for (int j = 0; j < 4; ++j)
        wsc[j] = wscale[(nt0 + j) * 16 + col0];

#pragma unroll
    for (int i = 0; i < 4; ++i) {
#pragma unroll
        for (int v = 0; v < 8; ++v) {
            const int row = (mt0 + i) * 16 + rowh + v;
            const float s = xs[row];
#pragma unroll
            for (int j = 0; j < 4; ++j) {
                const int col = (nt0 + j) * 16 + col0;
                out[(size_t)row * OUT_F + col] = (float)acc[i][j][v] * s * wsc[j];
            }
        }
    }
}

extern "C" void kernel_launch(void* const* d_in, const int* in_sizes, int n_in,
                              void* d_out, int out_size, void* d_ws, size_t ws_size,
                              hipStream_t stream) {
    (void)in_sizes; (void)n_in; (void)out_size; (void)ws_size;
    const float* x      = (const float*)d_in[0];
    const int*   wp     = (const int*)d_in[1];
    const float* wscale = (const float*)d_in[2];

    uint8_t* xq = (uint8_t*)d_ws;
    uint8_t* wq = xq + (size_t)N_TOKENS * IN_F;
    float*   xs = (float*)(wq + (size_t)OUT_F * IN_F);

    quant_x_kernel<<<N_TOKENS, 256, 0, stream>>>(x, xq, xs);
    unpack_w_kernel<<<OUT_F, 256, 0, stream>>>(wp, wq);

    const int grid = (N_TOKENS / 256) * (OUT_F / 256);
    gemm_w4a4_kernel<<<grid, 512, 0, stream>>>(
        xq, wq, xs, wscale, (float*)d_out);
}
